// NvidiaTransformerCore_32006096290241
// MI455X (gfx1250) — compile-verified, matrix operand order checked by reading, not run on hardware
//
#include <hip/hip_runtime.h>

// ---------------------------------------------------------------------------
// Transformer-XL forward (2 layers) for MI455X / gfx1250.
// All GEMMs run on v_wmma_f32_16x16x32_bf16 (bf16 in, f32 accumulate).
// Each wave computes a 32x64 output tile: 2 A-frags x 4 B-frags -> 8 WMMA
// per 32-deep K step (1.5 b128 loads per WMMA).  MWAVE template variant
// spreads waves across M (for narrow-N batched GEMMs like P*V, N=64).
// ---------------------------------------------------------------------------

typedef __attribute__((ext_vector_type(16))) __bf16 v16bf;
typedef __attribute__((ext_vector_type(8)))  float  v8f;

namespace {

constexpr int T_    = 1024;
constexpr int B_    = 4;
constexpr int D_    = 1024;
constexpr int H_    = 16;
constexpr int DH_   = 64;
constexpr int DI_   = 4096;
constexpr int L_    = 2;
constexpr int MLEN_ = 512;
constexpr int KLEN_ = MLEN_ + T_;        // 1536
constexpr long NEX_ = (long)T_ * B_ * D_; // 4,194,304

enum { GF_RELU = 1, GF_BIAS = 2, GF_OUTF32 = 4, GF_OUTBF16 = 8 };

// ----- helpers --------------------------------------------------------------

__device__ __forceinline__ unsigned short f2bf(float x) {
  union { float f; unsigned int u; } c; c.f = x;
  unsigned int u = c.u;
  unsigned int r = (u + 0x7FFFu + ((u >> 16) & 1u)) >> 16; // RNE
  return (unsigned short)r;
}

union Frag { v16bf v; unsigned int u[8]; };

// Load one 16x32 bf16 WMMA operand fragment (A row-major M x K, or B as the
// row-major N x K matrix whose transpose is multiplied).  Per ISA 7.12.2:
// lanes 0-15 hold row (lane&15) K 0..7/16..23, lanes 16-31 the +8 K-half.
__device__ __forceinline__ v16bf load_frag(const unsigned short* __restrict__ base,
                                           int ld, int row, int kBase, int lane) {
  Frag f;
  const unsigned short* rp = base + (long)row * ld + kBase + ((lane >> 4) << 3);
  f.u[0] = *(const unsigned int*)(rp + 0);
  f.u[1] = *(const unsigned int*)(rp + 2);
  f.u[2] = *(const unsigned int*)(rp + 4);
  f.u[3] = *(const unsigned int*)(rp + 6);
  f.u[4] = *(const unsigned int*)(rp + 16);
  f.u[5] = *(const unsigned int*)(rp + 18);
  f.u[6] = *(const unsigned int*)(rp + 20);
  f.u[7] = *(const unsigned int*)(rp + 22);
  return f.v;
}

__device__ __forceinline__ v8f wmma_bf16(v16bf a, v16bf b, v8f c) {
  return __builtin_amdgcn_wmma_f32_16x16x32_bf16(
      /*neg_a=*/false, a, /*neg_b=*/false, b,
      /*c_mod=*/(short)0, c, /*reuse_a=*/false, /*reuse_b=*/false);
}

// ----- generic batched WMMA GEMM:  C[z] = A[z] (MxK) * B[z] (NxK)^T ---------
// Batch z decomposed as (zo, zi) = (z / inner, z % inner); operand/base
// offsets are zo*Outer + zi*Inner (element units).
// MWAVE=false: block covers 32M x 256N (waves across N).  M%32, N%64, K%32.
//              Grid: x = ceil(N/256), y = M/32, z = batches.
// MWAVE=true:  block covers 128M x 64N  (waves across M; for narrow N).
//              Grid: x = ceil(N/64),  y = M/128, z = batches.  M%128==0.
// Block = 128 threads (4 waves).

template <bool MWAVE>
__global__ __launch_bounds__(128)
void gemm_wmma_kernel(const unsigned short* __restrict__ A,
                      const unsigned short* __restrict__ Bm,
                      const float* __restrict__ bias,
                      float* __restrict__ Cf,
                      unsigned short* __restrict__ Cb,
                      int M, int N, int K, int lda, int ldb, int ldc,
                      long aO, long aI, long bO, long bI, long cO, long cI,
                      int inner, int flags)
{
  const int lane = threadIdx.x & 31;
  const int wave = threadIdx.x >> 5;
  const int z    = blockIdx.z;
  const int zo   = z / inner;
  const int zi   = z - zo * inner;
  A  += zo * aO + zi * aI;
  Bm += zo * bO + zi * bI;
  const long cbase = zo * cO + zi * cI;

  int mBase, nBase;
  if (MWAVE) {
    mBase = (blockIdx.y * 4 + wave) * 32;
    nBase = blockIdx.x * 64;
  } else {
    mBase = blockIdx.y * 32;
    nBase = blockIdx.x * 256 + wave * 64;
  }
  if (nBase >= N || mBase >= M) return;   // wave-uniform

  const int arow = mBase + (lane & 15);
  const int brow = nBase + (lane & 15);

  v8f acc[2][4] = {};
  for (int k = 0; k < K; k += 32) {
    if (k + 128 < K) {
      __builtin_prefetch(A  + (long)arow * lda + k + 128, 0, 3);
      __builtin_prefetch(Bm + (long)brow * ldb + k + 128, 0, 3);
    }
    const v16bf fa0 = load_frag(A,  lda, arow,      k, lane);
    const v16bf fa1 = load_frag(A,  lda, arow + 16, k, lane);
    const v16bf fb0 = load_frag(Bm, ldb, brow,      k, lane);
    const v16bf fb1 = load_frag(Bm, ldb, brow + 16, k, lane);
    const v16bf fb2 = load_frag(Bm, ldb, brow + 32, k, lane);
    const v16bf fb3 = load_frag(Bm, ldb, brow + 48, k, lane);
    acc[0][0] = wmma_bf16(fa0, fb0, acc[0][0]);
    acc[0][1] = wmma_bf16(fa0, fb1, acc[0][1]);
    acc[0][2] = wmma_bf16(fa0, fb2, acc[0][2]);
    acc[0][3] = wmma_bf16(fa0, fb3, acc[0][3]);
    acc[1][0] = wmma_bf16(fa1, fb0, acc[1][0]);
    acc[1][1] = wmma_bf16(fa1, fb1, acc[1][1]);
    acc[1][2] = wmma_bf16(fa1, fb2, acc[1][2]);
    acc[1][3] = wmma_bf16(fa1, fb3, acc[1][3]);
  }

  const int mo = (lane >> 4) * 8;
#pragma unroll
  for (int ni = 0; ni < 4; ++ni) {
    const int   n  = nBase + ni * 16 + (lane & 15);
    const float bv = (flags & GF_BIAS) ? bias[n] : 0.0f;
#pragma unroll
    for (int mi = 0; mi < 2; ++mi) {
#pragma unroll
      for (int r = 0; r < 8; ++r) {
        const int m = mBase + mi * 16 + mo + r;
        float val = acc[mi][ni][r] + bv;
        if (flags & GF_RELU) val = fmaxf(val, 0.0f);
        const long idx = cbase + (long)m * ldc + n;
        if (flags & GF_OUTF32)  Cf[idx] = val;
        if (flags & GF_OUTBF16) Cb[idx] = f2bf(val);
      }
    }
  }
}

// ----- elementwise / pack kernels -------------------------------------------

__global__ void cvt_bf16_kernel(const float* __restrict__ s,
                                unsigned short* __restrict__ d, long n) {
  long i = (long)blockIdx.x * 256 + threadIdx.x;
  if (i < n) d[i] = f2bf(s[i]);
}

__global__ void copy_f32_kernel(const float* __restrict__ s,
                                float* __restrict__ d, long n) {
  long i = (long)blockIdx.x * 256 + threadIdx.x;
  if (i < n) d[i] = s[i];
}

// pos_emb (klen, D) bf16: [sin(p*invf) | cos(p*invf)], p = klen-1-row
__global__ void posemb_kernel(unsigned short* __restrict__ pos) {
  long i = (long)blockIdx.x * 256 + threadIdx.x;
  if (i >= (long)KLEN_ * D_) return;
  int row = (int)(i / D_);
  int c   = (int)(i - (long)row * D_);
  float p = (float)(KLEN_ - 1 - row);
  int   f = (c < D_ / 2) ? c : c - D_ / 2;
  float inv = expf(-((float)(2 * f) / (float)D_) * logf(10000.0f));
  float a = p * inv;
  pos[i] = f2bf((c < D_ / 2) ? sinf(a) : cosf(a));
}

// cat = [memory_l ; x]  as bf16 (klen*B, D)
__global__ void pack_cat_kernel(const float* __restrict__ mem,
                                const float* __restrict__ xf,
                                unsigned short* __restrict__ cat) {
  long i = (long)blockIdx.x * 256 + threadIdx.x;
  const long tot = (long)KLEN_ * B_ * D_;
  if (i >= tot) return;
  long j = i / (B_ * D_);
  long r = i - j * (B_ * D_);
  float v = (j < MLEN_) ? mem[i] : xf[(j - MLEN_) * (long)(B_ * D_) + r];
  cat[i] = f2bf(v);
}

// From w_heads (klen*B, 3D) f32 build:
//   qa,qb (B,H,T,Dh) bf16 (q + r_w_bias / q + r_r_bias)
//   kk    (B,H,klen,Dh) bf16
//   vt    (B,H,Dh,klen) bf16  (V transposed so P*V GEMM is K-contiguous)
__global__ void pack_qkv_kernel(const float* __restrict__ wh,
                                const float* __restrict__ rwb,
                                const float* __restrict__ rrb,
                                unsigned short* __restrict__ qa,
                                unsigned short* __restrict__ qb,
                                unsigned short* __restrict__ kk,
                                unsigned short* __restrict__ vt) {
  long i = (long)blockIdx.x * 256 + threadIdx.x;
  const long tot = (long)KLEN_ * B_ * D_;
  if (i >= tot) return;
  int j  = (int)(i / (B_ * D_));
  int r  = (int)(i - (long)j * (B_ * D_));
  int b  = r / D_;
  int hd = r - b * D_;
  int h  = hd / DH_;
  int d  = hd - h * DH_;
  const float* wrow = wh + ((long)j * B_ + b) * (3 * D_);
  long bh = (long)b * H_ + h;
  kk[(bh * KLEN_ + j) * DH_ + d] = f2bf(wrow[D_ + hd]);
  vt[(bh * DH_ + d) * KLEN_ + j] = f2bf(wrow[2 * D_ + hd]);
  if (j >= MLEN_) {
    int t = j - MLEN_;
    float q = wrow[hd];
    qa[(bh * T_ + t) * DH_ + d] = f2bf(q + rwb[hd]);
    qb[(bh * T_ + t) * DH_ + d] = f2bf(q + rrb[hd]);
  }
}

// rk (klen, D) f32 -> rb (H, klen, Dh) bf16
__global__ void pack_r_kernel(const float* __restrict__ rk,
                              unsigned short* __restrict__ rb) {
  long i = (long)blockIdx.x * 256 + threadIdx.x;
  if (i >= (long)KLEN_ * D_) return;
  int j  = (int)(i / D_);
  int hd = (int)(i - (long)j * D_);
  int h  = hd / DH_;
  int d  = hd - h * DH_;
  rb[((long)h * KLEN_ + j) * DH_ + d] = f2bf(rk[i]);
}

// score = scale*(AC + rel_shift(BD)); causal mask; softmax over j; bf16 prob.
// Faithful _rel_shift: y[b,h,i,j] = (n1==0) ? 0 : BDraw[b', n1-1, i, j]
// with q = B + H*b + h, b' = q/(H+1), n1 = q%(H+1).
__global__ __launch_bounds__(256)
void attn_softmax_kernel(const float* __restrict__ AC,
                         const float* __restrict__ BD,
                         unsigned short* __restrict__ prob) {
  constexpr int NPJ = KLEN_ / 256; // 6
  const int i = blockIdx.x;        // query position
  const int z = blockIdx.y;        // b*H + h
  const int b = z / H_;
  const int h = z - b * H_;
  const int q  = B_ + H_ * b + h;
  const int bp = q / (H_ + 1);
  const int n1 = q - bp * (H_ + 1);

  const float* acr = AC + ((long)z * T_ + i) * KLEN_;
  const float* bdr = (n1 == 0) ? nullptr
      : BD + (((long)bp * H_ + (n1 - 1)) * T_ + i) * KLEN_;
  const int jmax = i + MLEN_;
  const float scale = 0.125f; // 1/sqrt(64)

  __shared__ float red[256];

  float mx = -INFINITY;
  float sv[NPJ];
#pragma unroll
  for (int c = 0; c < NPJ; ++c) {
    int j = threadIdx.x + c * 256;
    float s = -INFINITY;
    if (j <= jmax) {
      s = (acr[j] + (bdr ? bdr[j] : 0.0f)) * scale;
      mx = fmaxf(mx, s);
    }
    sv[c] = s;
  }
  red[threadIdx.x] = mx; __syncthreads();
  for (int t = 128; t > 0; t >>= 1) {
    if (threadIdx.x < t) red[threadIdx.x] = fmaxf(red[threadIdx.x], red[threadIdx.x + t]);
    __syncthreads();
  }
  mx = red[0]; __syncthreads();

  float sum = 0.0f;
  float ev[NPJ];
#pragma unroll
  for (int c = 0; c < NPJ; ++c) {
    float e = (sv[c] == -INFINITY) ? 0.0f : __expf(sv[c] - mx);
    ev[c] = e; sum += e;
  }
  red[threadIdx.x] = sum; __syncthreads();
  for (int t = 128; t > 0; t >>= 1) {
    if (threadIdx.x < t) red[threadIdx.x] += red[threadIdx.x + t];
    __syncthreads();
  }
  const float invs = 1.0f / red[0];

  unsigned short* pr = prob + ((long)z * T_ + i) * KLEN_;
#pragma unroll
  for (int c = 0; c < NPJ; ++c) {
    int j = threadIdx.x + c * 256;
    pr[j] = f2bf(ev[c] * invs);
  }
}

// out = LayerNorm(resid + delta) * g + be ; optional bf16 copy. In-place safe.
__global__ __launch_bounds__(256)
void residual_ln_kernel(const float* __restrict__ resid,
                        const float* __restrict__ delta,
                        const float* __restrict__ g,
                        const float* __restrict__ be,
                        float* __restrict__ outf,
                        unsigned short* __restrict__ outb) {
  constexpr int NPC = D_ / 256; // 4
  __shared__ float red[256];
  const long row = blockIdx.x;
  const float* rp = resid + row * D_;
  const float* dp = delta ? delta + row * D_ : nullptr;

  float v[NPC];
  float s = 0.0f;
#pragma unroll
  for (int c = 0; c < NPC; ++c) {
    int jj = threadIdx.x + c * 256;
    float x = rp[jj] + (dp ? dp[jj] : 0.0f);
    v[c] = x; s += x;
  }
  red[threadIdx.x] = s; __syncthreads();
  for (int t = 128; t > 0; t >>= 1) {
    if (threadIdx.x < t) red[threadIdx.x] += red[threadIdx.x + t];
    __syncthreads();
  }
  const float mean = red[0] / (float)D_; __syncthreads();

  float s2 = 0.0f;
#pragma unroll
  for (int c = 0; c < NPC; ++c) { float d = v[c] - mean; s2 += d * d; }
  red[threadIdx.x] = s2; __syncthreads();
  for (int t = 128; t > 0; t >>= 1) {
    if (threadIdx.x < t) red[threadIdx.x] += red[threadIdx.x + t];
    __syncthreads();
  }
  const float inv = rsqrtf(red[0] / (float)D_ + 1e-5f);

#pragma unroll
  for (int c = 0; c < NPC; ++c) {
    int jj = threadIdx.x + c * 256;
    float y = (v[c] - mean) * inv * g[jj] + be[jj];
    outf[row * D_ + jj] = y;
    if (outb) outb[row * D_ + jj] = f2bf(y);
  }
}

// ----- host-side launch helpers ---------------------------------------------

inline dim3 g1(long n) { return dim3((unsigned)((n + 255) / 256)); }

inline void gemm(hipStream_t st, const unsigned short* A, const unsigned short* B,
                 const float* bias, float* Cf, unsigned short* Cb,
                 int M, int N, int K, int lda, int ldb, int ldc,
                 long aO, long aI, long bO, long bI, long cO, long cI,
                 int inner, int nb, int flags, bool mwave = false) {
  if (mwave) {
    dim3 grid((unsigned)((N + 63) / 64), (unsigned)(M / 128), (unsigned)nb);
    gemm_wmma_kernel<true><<<grid, dim3(128), 0, st>>>(A, B, bias, Cf, Cb,
        M, N, K, lda, ldb, ldc, aO, aI, bO, bI, cO, cI, inner, flags);
  } else {
    dim3 grid((unsigned)((N + 255) / 256), (unsigned)(M / 32), (unsigned)nb);
    gemm_wmma_kernel<false><<<grid, dim3(128), 0, st>>>(A, B, bias, Cf, Cb,
        M, N, K, lda, ldb, ldc, aO, aI, bO, bI, cO, cI, inner, flags);
  }
}

} // namespace

// ---------------------------------------------------------------------------

extern "C" void kernel_launch(void* const* d_in, const int* in_sizes, int n_in,
                              void* d_out, int out_size, void* d_ws, size_t ws_size,
                              hipStream_t stream) {
  (void)in_sizes; (void)n_in; (void)out_size; (void)ws_size;

  const float* inputs   = (const float*)d_in[0];
  const float* memory   = (const float*)d_in[1];
  const float* qkv_w    = (const float*)d_in[2];
  const float* o_w      = (const float*)d_in[3];
  const float* r_w      = (const float*)d_in[4];
  const float* ln1_g    = (const float*)d_in[5];
  const float* ln1_b    = (const float*)d_in[6];
  const float* ff_w1    = (const float*)d_in[7];
  const float* ff_b1    = (const float*)d_in[8];
  const float* ff_w2    = (const float*)d_in[9];
  const float* ff_b2    = (const float*)d_in[10];
  const float* ln2_g    = (const float*)d_in[11];
  const float* ln2_b    = (const float*)d_in[12];
  const float* r_w_bias = (const float*)d_in[13];
  const float* r_r_bias = (const float*)d_in[14];
  const float* fin_g    = (const float*)d_in[15];
  const float* fin_b    = (const float*)d_in[16];

  char* ws = (char*)d_ws;
  size_t off = 0;
  auto take = [&](size_t bytes) -> char* {
    char* p = ws + off;
    off = (off + bytes + 255) & ~(size_t)255;
    return p;
  };

  float*          xf   = (float*)take(NEX_ * 4);
  unsigned short* xbf  = (unsigned short*)take(NEX_ * 2);
  unsigned short* cat  = (unsigned short*)take((size_t)KLEN_ * B_ * D_ * 2);
  float*          wh   = (float*)take((size_t)KLEN_ * B_ * 3 * D_ * 4);
  unsigned short* qa   = (unsigned short*)take(NEX_ * 2);
  unsigned short* qb   = (unsigned short*)take(NEX_ * 2);
  unsigned short* kk   = (unsigned short*)take((size_t)KLEN_ * B_ * D_ * 2);
  unsigned short* vt   = (unsigned short*)take((size_t)KLEN_ * B_ * D_ * 2);
  float*          rk   = (float*)take((size_t)KLEN_ * D_ * 4);
  unsigned short* rb   = (unsigned short*)take((size_t)KLEN_ * D_ * 2);
  float*          AC   = (float*)take((size_t)B_ * H_ * T_ * KLEN_ * 4);
  float*          BDm  = (float*)take((size_t)B_ * H_ * T_ * KLEN_ * 4);
  unsigned short* prob = (unsigned short*)take((size_t)B_ * H_ * T_ * KLEN_ * 2);
  unsigned short* vecb = (unsigned short*)take(NEX_ * 2);
  float*          dlt  = (float*)take(NEX_ * 4);
  unsigned short* hbf  = (unsigned short*)take((size_t)T_ * B_ * DI_ * 2);
  unsigned short* wq   = (unsigned short*)take((size_t)L_ * 3 * D_ * D_ * 2);
  unsigned short* wo   = (unsigned short*)take((size_t)L_ * D_ * D_ * 2);
  unsigned short* wr   = (unsigned short*)take((size_t)L_ * D_ * D_ * 2);
  unsigned short* w1   = (unsigned short*)take((size_t)L_ * DI_ * D_ * 2);
  unsigned short* w2   = (unsigned short*)take((size_t)L_ * D_ * DI_ * 2);
  unsigned short* pos  = (unsigned short*)take((size_t)KLEN_ * D_ * 2);

  // ---- once-per-call prep --------------------------------------------------
  hipMemcpyAsync(xf, inputs, NEX_ * sizeof(float), hipMemcpyDeviceToDevice, stream);
  posemb_kernel<<<g1((long)KLEN_ * D_), 256, 0, stream>>>(pos);
  cvt_bf16_kernel<<<g1((long)L_ * 3 * D_ * D_), 256, 0, stream>>>(qkv_w, wq, (long)L_ * 3 * D_ * D_);
  cvt_bf16_kernel<<<g1((long)L_ * D_ * D_),     256, 0, stream>>>(o_w,   wo, (long)L_ * D_ * D_);
  cvt_bf16_kernel<<<g1((long)L_ * D_ * D_),     256, 0, stream>>>(r_w,   wr, (long)L_ * D_ * D_);
  cvt_bf16_kernel<<<g1((long)L_ * DI_ * D_),    256, 0, stream>>>(ff_w1, w1, (long)L_ * DI_ * D_);
  cvt_bf16_kernel<<<g1((long)L_ * D_ * DI_),    256, 0, stream>>>(ff_w2, w2, (long)L_ * D_ * DI_);

  for (int l = 0; l < L_; ++l) {
    const float* mem_l = memory + (size_t)l * MLEN_ * B_ * D_;

    // cat = [mem_l ; x]  (bf16)
    pack_cat_kernel<<<g1((long)KLEN_ * B_ * D_), 256, 0, stream>>>(mem_l, xf, cat);

    // w_heads = cat @ qkv_w[l].T   (klen*B, 3D)
    gemm(stream, cat, wq + (size_t)l * 3 * D_ * D_, nullptr, wh, nullptr,
         KLEN_ * B_, 3 * D_, D_, D_, D_, 3 * D_,
         0, 0, 0, 0, 0, 0, 1, 1, GF_OUTF32);

    // r_head_k = pos_emb @ r_w[l].T  (klen, D)
    gemm(stream, pos, wr + (size_t)l * D_ * D_, nullptr, rk, nullptr,
         KLEN_, D_, D_, D_, D_, D_,
         0, 0, 0, 0, 0, 0, 1, 1, GF_OUTF32);

    pack_qkv_kernel<<<g1((long)KLEN_ * B_ * D_), 256, 0, stream>>>(
        wh, r_w_bias, r_r_bias, qa, qb, kk, vt);
    pack_r_kernel<<<g1((long)KLEN_ * D_), 256, 0, stream>>>(rk, rb);

    // AC[b,h] = (q + r_w_bias) @ k^T   batched over b*H
    gemm(stream, qa, kk, nullptr, AC, nullptr,
         T_, KLEN_, DH_, DH_, DH_, KLEN_,
         0, (long)T_ * DH_, 0, (long)KLEN_ * DH_, 0, (long)T_ * KLEN_,
         B_ * H_, B_ * H_, GF_OUTF32);

    // BDraw[b,h] = (q + r_r_bias) @ r_head_k[h]^T
    gemm(stream, qb, rb, nullptr, BDm, nullptr,
         T_, KLEN_, DH_, DH_, DH_, KLEN_,
         (long)H_ * T_ * DH_, (long)T_ * DH_,
         0, (long)KLEN_ * DH_,
         (long)H_ * T_ * KLEN_, (long)T_ * KLEN_,
         H_, B_ * H_, GF_OUTF32);

    // rel-shift + mask + softmax -> prob (bf16)
    attn_softmax_kernel<<<dim3(T_, B_ * H_), 256, 0, stream>>>(AC, BDm, prob);

    // vec[b,h] = prob @ v  -> stored directly as (T,B,D) bf16
    // Narrow N (=64): use MWAVE variant so all 4 waves per block are active.
    gemm(stream, prob, vt, nullptr, nullptr, vecb,
         T_, DH_, KLEN_, KLEN_, KLEN_, B_ * D_,
         (long)H_ * T_ * KLEN_, (long)T_ * KLEN_,
         (long)H_ * DH_ * KLEN_, (long)DH_ * KLEN_,
         (long)D_, (long)DH_,
         H_, B_ * H_, GF_OUTBF16, /*mwave=*/true);

    // attn_out = vec @ o_w[l].T
    gemm(stream, vecb, wo + (size_t)l * D_ * D_, nullptr, dlt, nullptr,
         T_ * B_, D_, D_, D_, D_, D_,
         0, 0, 0, 0, 0, 0, 1, 1, GF_OUTF32);

    // x = LN(x + attn_out)
    residual_ln_kernel<<<T_ * B_, 256, 0, stream>>>(
        xf, dlt, ln1_g + (size_t)l * D_, ln1_b + (size_t)l * D_, xf, xbf);

    // h = relu(x @ w1^T + b1)  (bf16)
    gemm(stream, xbf, w1 + (size_t)l * DI_ * D_, ff_b1 + (size_t)l * DI_, nullptr, hbf,
         T_ * B_, DI_, D_, D_, D_, DI_,
         0, 0, 0, 0, 0, 0, 1, 1, GF_BIAS | GF_RELU | GF_OUTBF16);

    // ff = h @ w2^T + b2
    gemm(stream, hbf, w2 + (size_t)l * D_ * DI_, ff_b2 + (size_t)l * D_, dlt, nullptr,
         T_ * B_, D_, DI_, DI_, DI_, D_,
         0, 0, 0, 0, 0, 0, 1, 1, GF_BIAS | GF_OUTF32);

    // x = LN(x + ff)
    residual_ln_kernel<<<T_ * B_, 256, 0, stream>>>(
        xf, dlt, ln2_g + (size_t)l * D_, ln2_b + (size_t)l * D_, xf, xbf);

    // new_mem[l] = hids[l][T-mlen:]
    copy_f32_kernel<<<g1((long)MLEN_ * B_ * D_), 256, 0, stream>>>(
        xf + (size_t)(T_ - MLEN_) * B_ * D_,
        (float*)d_out + NEX_ + (size_t)l * MLEN_ * B_ * D_,
        (long)MLEN_ * B_ * D_);
  }

  // out = LN(x) with final gamma/beta
  residual_ln_kernel<<<T_ * B_, 256, 0, stream>>>(
      xf, nullptr, fin_g, fin_b, (float*)d_out, nullptr);
}
